// ForwardPolicy_56925496541574
// MI455X (gfx1250) — matrix operand order repaired and checked by reading, not compiled, not run on hardware
//
#include <hip/hip_runtime.h>
#include <hip/hip_bf16.h>

typedef __attribute__((ext_vector_type(16))) _Float16 v16h;
typedef __attribute__((ext_vector_type(8)))  float    v8f;

// ---------- helpers ----------
__device__ __forceinline__ unsigned encf(float x) {
    unsigned u = __float_as_uint(x);
    return (u & 0x80000000u) ? ~u : (u | 0x80000000u);   // monotone map float -> u32
}
__device__ __forceinline__ float decf(unsigned e) {
    unsigned u = (e >> 31) ? (e ^ 0x80000000u) : ~e;
    return __uint_as_float(u);
}

// ---------- fills / init ----------
__global__ void fill_f32(float* p, float v, size_t n) {
    size_t i = (size_t)blockIdx.x * 256 + threadIdx.x;
    if (i < n) p[i] = v;
}
__global__ void fill_u32(unsigned* p, unsigned v, size_t n) {
    size_t i = (size_t)blockIdx.x * 256 + threadIdx.x;
    if (i < n) p[i] = v;
}
__global__ void fill_i32(int* p, int v, size_t n) {
    size_t i = (size_t)blockIdx.x * 256 + threadIdx.x;
    if (i < n) p[i] = v;
}
__global__ void fill_bias(float* p, const float* __restrict__ b, int ncols, size_t n) {
    size_t i = (size_t)blockIdx.x * 256 + threadIdx.x;
    if (i < n) p[i] = b[i % ncols];
}
__global__ void mask_scatter(int* amask, const int* __restrict__ actions, int k) {
    int i = blockIdx.x * 256 + threadIdx.x;
    if (i < k) amask[actions[i]] = 0;
}

// ---------- self-loop edge_attr mean ----------
__global__ void edge_count(const int* __restrict__ ei, const float* __restrict__ ea,
                           float* cnt, float* sumea, int E) {
    int e = blockIdx.x * 256 + threadIdx.x;
    if (e >= E) return;
    int dst = ei[E + e];
    atomicAdd(&cnt[dst], 1.0f);
    atomicAdd(&sumea[dst], ea[e]);
}
__global__ void mean_kernel(const float* __restrict__ cnt, const float* __restrict__ sumea,
                            float* mean, int N) {
    int n = blockIdx.x * 256 + threadIdx.x;
    if (n < N) mean[n] = sumea[n] / fmaxf(cnt[n], 1.0f);
}

// ---------- dual-output WMMA GEMM:  O = (reluA? relu(A) : A)[M,K] @ W[N,K]^T + b ----------
// One 16x16 output tile per wave; two weight matrices share the A fragment.
__global__ void gemm_dual_wmma(const float* __restrict__ A, int M, int K, int NC, int reluA,
                               const float* __restrict__ W0, const float* __restrict__ b0, float* __restrict__ O0,
                               const float* __restrict__ W1, const float* __restrict__ b1, float* __restrict__ O1) {
    const int lane  = threadIdx.x & 31;
    const int wave  = threadIdx.x >> 5;
    const int mtile = blockIdx.x * 8 + wave;
    const int ntile = blockIdx.y;
    if (mtile * 16 >= M) return;
    const int h = lane >> 4;   // half-wave
    const int m = lane & 15;   // A row within tile == B/C column within tile

    v8f acc0 = {}; v8f acc1 = {};
    const float* arow  = A  + (size_t)(mtile * 16 + m) * K;
    const float* w0row = W0 + (size_t)(ntile * 16 + m) * K;
    const float* w1row = W1 + (size_t)(ntile * 16 + m) * K;

    for (int kk = 0; kk < K; kk += 32) {
        v16h a, bb0, bb1;
        // A 16x32 f16 layout: VGPR j<4: K=8h+2j+p ; j>=4: K=16+8h+2(j-4)+p
#pragma unroll
        for (int i = 0; i < 16; ++i) {
            int j = i >> 1, p = i & 1;
            int kl = (j < 4) ? (8 * h + 2 * j + p) : (16 + 8 * h + 2 * (j - 4) + p);
            float av = arow[kk + kl];
            if (reluA) av = fmaxf(av, 0.f);
            a[i] = (_Float16)av;
        }
        // B 32x16 f16 layout: same K order as A: i<8: K=8h+i ; i>=8: K=16+8h+(i-8)
#pragma unroll
        for (int i = 0; i < 16; ++i) {
            int kb = (i < 8) ? (8 * h + i) : (16 + 8 * h + (i - 8));
            bb0[i] = (_Float16)w0row[kk + kb];
            bb1[i] = (_Float16)w1row[kk + kb];
        }
        acc0 = __builtin_amdgcn_wmma_f32_16x16x32_f16(false, a, false, bb0, (short)0, acc0, false, false);
        acc1 = __builtin_amdgcn_wmma_f32_16x16x32_f16(false, a, false, bb1, (short)0, acc1, false, false);
    }
    const int col = ntile * 16 + m;
    const float bv0 = b0[col], bv1 = b1[col];
#pragma unroll
    for (int r = 0; r < 8; ++r) {
        int row = mtile * 16 + r + 8 * h;   // C/D: VGPR r -> M=r (lanes 0-15), M=r+8 (lanes 16-31)
        O0[(size_t)row * NC + col] = acc0[r] + bv0;
        O1[(size_t)row * NC + col] = acc1[r] + bv1;
    }
}

// ---------- layer-1 edge phase (H=4, C=64, HC=256); wave per edge, 8 ch/lane ----------
__global__ void score1_kernel(const float* __restrict__ xl, const float* __restrict__ xr,
                              const int* __restrict__ ei, const float* __restrict__ ea,
                              const float* __restrict__ mean_ea, const float* __restrict__ We,
                              const float* __restrict__ att, float* s, unsigned* smax, int E, int N) {
    int e = blockIdx.x * 8 + (threadIdx.x >> 5);
    int lane = threadIdx.x & 31;
    if (e >= E + N) return;
    int src, dst; float eav;
    if (e < E) { src = ei[e]; dst = ei[E + e]; eav = ea[e]; }
    else       { src = dst = e - E; eav = mean_ea[src]; }
    const float* pl = xl + (size_t)src * 256;
    const float* pr = xr + (size_t)dst * 256;
    int c0 = lane * 8;
    float part = 0.f;
#pragma unroll
    for (int i = 0; i < 8; ++i) {
        int c = c0 + i;
        float v = pl[c] + pr[c] + eav * We[c];
        v = v > 0.f ? v : 0.2f * v;           // leaky_relu(0.2)
        part += v * att[c];
    }
    part += __shfl_xor(part, 1, 32);
    part += __shfl_xor(part, 2, 32);
    part += __shfl_xor(part, 4, 32);
    if ((lane & 7) == 0) {
        int head = lane >> 3;
        s[(size_t)e * 4 + head] = part;
        atomicMax(&smax[dst * 4 + head], encf(part));
    }
}
__global__ void expden1(float* s, const int* __restrict__ ei, const unsigned* __restrict__ smax,
                        float* den, int E, int N) {
    int idx = blockIdx.x * 256 + threadIdx.x;
    if (idx >= (E + N) * 4) return;
    int e = idx >> 2, hh = idx & 3;
    int dst = (e < E) ? ei[E + e] : (e - E);
    float ex = expf(s[idx] - decf(smax[dst * 4 + hh]));
    s[idx] = ex;
    atomicAdd(&den[dst * 4 + hh], ex);
}
__global__ void agg1(const float* __restrict__ xl, const float* __restrict__ s,
                     const float* __restrict__ den, const int* __restrict__ ei,
                     float* out, int E, int N) {
    int e = blockIdx.x * 8 + (threadIdx.x >> 5);
    int lane = threadIdx.x & 31;
    if (e >= E + N) return;
    int src, dst;
    if (e < E) { src = ei[e]; dst = ei[E + e]; } else { src = dst = e - E; }
    int head = lane >> 3;
    float a = s[(size_t)e * 4 + head] / (den[dst * 4 + head] + 1e-16f);
    const float* pl = xl + (size_t)src * 256;
    float* po = out + (size_t)dst * 256;
    int c0 = lane * 8;
#pragma unroll
    for (int i = 0; i < 8; ++i) atomicAdd(&po[c0 + i], a * pl[c0 + i]);
}

// ---------- layer-2 edge phase (H=1, C=64); wave per edge, 2 ch/lane ----------
__global__ void score2_kernel(const float* __restrict__ xl, const float* __restrict__ xr,
                              const int* __restrict__ ei, const float* __restrict__ ea,
                              const float* __restrict__ mean_ea, const float* __restrict__ We,
                              const float* __restrict__ att, float* s, unsigned* smax, int E, int N) {
    int e = blockIdx.x * 8 + (threadIdx.x >> 5);
    int lane = threadIdx.x & 31;
    if (e >= E + N) return;
    int src, dst; float eav;
    if (e < E) { src = ei[e]; dst = ei[E + e]; eav = ea[e]; }
    else       { src = dst = e - E; eav = mean_ea[src]; }
    const float* pl = xl + (size_t)src * 64;
    const float* pr = xr + (size_t)dst * 64;
    float part = 0.f;
#pragma unroll
    for (int q = 0; q < 2; ++q) {
        int c = lane + q * 32;
        float v = pl[c] + pr[c] + eav * We[c];
        v = v > 0.f ? v : 0.2f * v;
        part += v * att[c];
    }
    for (int off = 16; off; off >>= 1) part += __shfl_xor(part, off, 32);
    if (lane == 0) { s[e] = part; atomicMax(&smax[dst], encf(part)); }
}
__global__ void expden2(float* s, const int* __restrict__ ei, const unsigned* __restrict__ smax,
                        float* den, int E, int N) {
    int e = blockIdx.x * 256 + threadIdx.x;
    if (e >= E + N) return;
    int dst = (e < E) ? ei[E + e] : (e - E);
    float ex = expf(s[e] - decf(smax[dst]));
    s[e] = ex;
    atomicAdd(&den[dst], ex);
}
__global__ void agg2(const float* __restrict__ xl, const float* __restrict__ s,
                     const float* __restrict__ den, const int* __restrict__ ei,
                     float* out, int E, int N) {
    int e = blockIdx.x * 8 + (threadIdx.x >> 5);
    int lane = threadIdx.x & 31;
    if (e >= E + N) return;
    int src, dst;
    if (e < E) { src = ei[e]; dst = ei[E + e]; } else { src = dst = e - E; }
    float a = s[e] / (den[dst] + 1e-16f);
    const float* pl = xl + (size_t)src * 64;
    float* po = out + (size_t)dst * 64;
    atomicAdd(&po[lane],      a * pl[lane]);
    atomicAdd(&po[lane + 32], a * pl[lane + 32]);
}

// ---------- mean pool (relu folded) + potential head + sigmoid(alpha) ----------
__global__ void pool_partial(const float* __restrict__ h2, int N, float* gsum) {
    __shared__ float acc[256];
    int t = threadIdx.x, c = t & 63, r = t >> 6;
    float local = 0.f;
    for (int n = blockIdx.x * 4 + r; n < N; n += gridDim.x * 4)
        local += fmaxf(h2[(size_t)n * 64 + c], 0.f);          // relu(h2) folded here
    acc[t] = local;
    __syncthreads();
    if (t < 64) atomicAdd(&gsum[c], acc[t] + acc[t + 64] + acc[t + 128] + acc[t + 192]);
}
__global__ void head_kernel(const float* __restrict__ gsum, int N, const float* __restrict__ Wp,
                            const float* __restrict__ bp, const float* __restrict__ alpha,
                            float* g, float* out, int num_actions) {
    __shared__ float sh[64];
    int t = threadIdx.x;                       // 64 threads
    float gv = gsum[t] / (float)N;
    g[t] = gv;
    sh[t] = gv * Wp[t];
    __syncthreads();
    for (int off = 32; off; off >>= 1) { if (t < off) sh[t] += sh[t + off]; __syncthreads(); }
    if (t == 0) {
        float d = sh[0] + bp[0];
        out[num_actions]     = d > 0.f ? d : 0.f;                 // state_potential
        out[num_actions + 1] = 1.f / (1.f + expf(-alpha[0]));     // sigmoid(alpha)
    }
}

// ---------- masked logits (wave per action, coalesced 256B rows of Wf) ----------
__global__ void logits_kernel(const float* __restrict__ g, const float* __restrict__ Wf,
                              const float* __restrict__ bf, const int* __restrict__ ei,
                              const int* __restrict__ amask, float* logits,
                              unsigned* maxEnc, int E, int num_actions) {
    __shared__ float gs[64];
    if (threadIdx.x < 64) gs[threadIdx.x] = g[threadIdx.x];
    __syncthreads();
    int j = blockIdx.x * 8 + (threadIdx.x >> 5);
    int lane = threadIdx.x & 31;
    if (j >= num_actions) return;
    bool ok = amask[j] != 0;
    if (j < E) ok = ok && (ei[j] != ei[E + j]);
    float val = -__builtin_inff();
    if (ok) {
        const float* wr = Wf + (size_t)j * 64;
        __builtin_prefetch(Wf + (size_t)(j + 8) * 64, 0, 0);   // global_prefetch_b8
        float part = gs[lane] * wr[lane] + gs[lane + 32] * wr[lane + 32];
        for (int off = 16; off; off >>= 1) part += __shfl_xor(part, off, 32);
        val = part + bf[j];
    }
    if (lane == 0) {
        logits[j] = val;
        if (ok) atomicMax(maxEnc, encf(val));
    }
}
__global__ void softmax_exp(const float* __restrict__ logits, const unsigned* __restrict__ maxEnc,
                            float* out, float* sumExp, int n) {
    __shared__ float sh[256];
    int j = blockIdx.x * 256 + threadIdx.x;
    float m = decf(*maxEnc);
    float ex = 0.f;
    if (j < n) {
        float v = logits[j];
        ex = (v == -__builtin_inff()) ? 0.f : expf(v - m);
        out[j] = ex;
    }
    sh[threadIdx.x] = ex;
    __syncthreads();
    for (int off = 128; off; off >>= 1) { if (threadIdx.x < off) sh[threadIdx.x] += sh[threadIdx.x + off]; __syncthreads(); }
    if (threadIdx.x == 0) atomicAdd(sumExp, sh[0]);
}
__global__ void softmax_norm(float* out, const float* __restrict__ sumExp, int n) {
    int j = blockIdx.x * 256 + threadIdx.x;
    if (j < n) out[j] = out[j] / (*sumExp);
}

// ---------------------------------------------------------------------------
extern "C" void kernel_launch(void* const* d_in, const int* in_sizes, int n_in,
                              void* d_out, int out_size, void* d_ws, size_t ws_size,
                              hipStream_t stream) {
    const float* x    = (const float*)d_in[0];
    const int*   ei   = (const int*)  d_in[1];
    const float* ea   = (const float*)d_in[2];
    const int*   acts = (const int*)  d_in[3];
    const float* Wl1  = (const float*)d_in[4];  const float* bl1 = (const float*)d_in[5];
    const float* Wr1  = (const float*)d_in[6];  const float* br1 = (const float*)d_in[7];
    const float* We1  = (const float*)d_in[8];  const float* att1= (const float*)d_in[9];
    const float* bias1= (const float*)d_in[10];
    const float* Wl2  = (const float*)d_in[11]; const float* bl2 = (const float*)d_in[12];
    const float* Wr2  = (const float*)d_in[13]; const float* br2 = (const float*)d_in[14];
    const float* We2  = (const float*)d_in[15]; const float* att2= (const float*)d_in[16];
    const float* bias2= (const float*)d_in[17];
    const float* Wf   = (const float*)d_in[18]; const float* bf  = (const float*)d_in[19];
    const float* Wp   = (const float*)d_in[20]; const float* bp  = (const float*)d_in[21];
    const float* alpha= (const float*)d_in[22];

    const int HC  = in_sizes[5];            // 256 = H*hid
    const int F   = in_sizes[4] / HC;       // 128
    const int N   = in_sizes[0] / F;        // 20000
    const int E   = in_sizes[2];            // 400000
    const int hid = in_sizes[12];           // 64
    const int E2  = E + N;
    const int numA = E + 1;                 // 400001
    const int nAct = in_sizes[3];
    float* out = (float*)d_out;

    // workspace carve-out (all fits in ~90 MB; L2-resident)
    char* w = (char*)d_ws;
    auto alloc = [&](size_t bytes) -> void* {
        void* p = (void*)w;
        w += (bytes + 255) & ~(size_t)255;
        return p;
    };
    float*    xl1    = (float*)   alloc((size_t)N * HC * 4);
    float*    xr1    = (float*)   alloc((size_t)N * HC * 4);
    float*    h1     = (float*)   alloc((size_t)N * HC * 4);
    float*    xl2    = (float*)   alloc((size_t)N * hid * 4);
    float*    xr2    = (float*)   alloc((size_t)N * hid * 4);
    float*    h2     = (float*)   alloc((size_t)N * hid * 4);
    float*    cnt    = (float*)   alloc((size_t)N * 4);
    float*    sumea  = (float*)   alloc((size_t)N * 4);
    float*    meanea = (float*)   alloc((size_t)N * 4);
    float*    s1     = (float*)   alloc((size_t)E2 * 4 * 4);
    unsigned* smax1  = (unsigned*)alloc((size_t)N * 4 * 4);
    float*    den1   = (float*)   alloc((size_t)N * 4 * 4);
    float*    s2     = (float*)   alloc((size_t)E2 * 4);
    unsigned* smax2  = (unsigned*)alloc((size_t)N * 4);
    float*    den2   = (float*)   alloc((size_t)N * 4);
    float*    gsum   = (float*)   alloc(64 * 4);
    float*    g      = (float*)   alloc(64 * 4);
    float*    logits = (float*)   alloc((size_t)numA * 4);
    int*      amask  = (int*)     alloc((size_t)numA * 4);
    unsigned* maxEnc = (unsigned*)alloc(256);
    float*    sumExp = (float*)   alloc(256);

    const unsigned NEG_INF_ENC = 0x007FFFFFu;   // encf(-inf)
    auto gs = [](size_t n) { return (unsigned)((n + 255) / 256); };

    // ---- init ----
    fill_bias<<<gs((size_t)N * HC), 256, 0, stream>>>(h1, bias1, HC, (size_t)N * HC);
    fill_bias<<<gs((size_t)N * hid), 256, 0, stream>>>(h2, bias2, hid, (size_t)N * hid);
    fill_f32<<<gs(N), 256, 0, stream>>>(cnt, 0.f, N);
    fill_f32<<<gs(N), 256, 0, stream>>>(sumea, 0.f, N);
    fill_u32<<<gs((size_t)N * 4), 256, 0, stream>>>(smax1, NEG_INF_ENC, (size_t)N * 4);
    fill_f32<<<gs((size_t)N * 4), 256, 0, stream>>>(den1, 0.f, (size_t)N * 4);
    fill_u32<<<gs(N), 256, 0, stream>>>(smax2, NEG_INF_ENC, N);
    fill_f32<<<gs(N), 256, 0, stream>>>(den2, 0.f, N);
    fill_f32<<<1, 256, 0, stream>>>(gsum, 0.f, 64);
    fill_i32<<<gs(numA), 256, 0, stream>>>(amask, 1, numA);
    fill_u32<<<1, 256, 0, stream>>>(maxEnc, NEG_INF_ENC, 1);
    fill_f32<<<1, 256, 0, stream>>>(sumExp, 0.f, 1);

    // ---- self-loop edge_attr mean ----
    edge_count<<<gs(E), 256, 0, stream>>>(ei, ea, cnt, sumea, E);
    mean_kernel<<<gs(N), 256, 0, stream>>>(cnt, sumea, meanea, N);

    // ---- layer 1 ----
    {
        int Mtiles = (N + 15) / 16;
        dim3 grid((Mtiles + 7) / 8, HC / 16);
        gemm_dual_wmma<<<grid, 256, 0, stream>>>(x, N, F, HC, 0, Wl1, bl1, xl1, Wr1, br1, xr1);
    }
    score1_kernel<<<(E2 + 7) / 8, 256, 0, stream>>>(xl1, xr1, ei, ea, meanea, We1, att1, s1, smax1, E, N);
    expden1<<<gs((size_t)E2 * 4), 256, 0, stream>>>(s1, ei, smax1, den1, E, N);
    agg1<<<(E2 + 7) / 8, 256, 0, stream>>>(xl1, s1, den1, ei, h1, E, N);

    // ---- layer 2 (relu(h1) folded into GEMM A-load) ----
    {
        int Mtiles = (N + 15) / 16;
        dim3 grid((Mtiles + 7) / 8, hid / 16);
        gemm_dual_wmma<<<grid, 256, 0, stream>>>(h1, N, HC, hid, 1, Wl2, bl2, xl2, Wr2, br2, xr2);
    }
    score2_kernel<<<(E2 + 7) / 8, 256, 0, stream>>>(xl2, xr2, ei, ea, meanea, We2, att2, s2, smax2, E, N);
    expden2<<<gs(E2), 256, 0, stream>>>(s2, ei, smax2, den2, E, N);
    agg2<<<(E2 + 7) / 8, 256, 0, stream>>>(xl2, s2, den2, ei, h2, E, N);

    // ---- pool (relu(h2) folded) + heads ----
    pool_partial<<<64, 256, 0, stream>>>(h2, N, gsum);
    head_kernel<<<1, 64, 0, stream>>>(gsum, N, Wp, bp, alpha, g, out, numA);

    // ---- masked softmax over actions ----
    mask_scatter<<<gs(nAct), 256, 0, stream>>>(amask, acts, nAct);
    logits_kernel<<<(numA + 7) / 8, 256, 0, stream>>>(g, Wf, bf, ei, amask, logits, maxEnc, E, numA);
    softmax_exp<<<gs(numA), 256, 0, stream>>>(logits, maxEnc, out, sumExp, numA);
    softmax_norm<<<gs(numA), 256, 0, stream>>>(out, sumExp, numA);
}
